// Attention_12816182412017
// MI455X (gfx1250) — matrix operand order repaired and checked by reading, not compiled, not run on hardware
//
#include <hip/hip_runtime.h>
#include <hip/hip_bf16.h>

// ---------------------------------------------------------------------------
// MHA forward for MI455X (gfx1250), wave32 + v_wmma_f32_16x16x32_f16.
// Pipeline:
//   0) pre-convert Wq/Wk/Wv/Wo f32 -> f16 (one streaming pass, 2 MB)
//   1) Q/K/V projections (f32 activations cvt on the fly, f16 weights),
//      outputs f16 in head-split layouts (V transposed for P@V B-frags)
//   2) attention: Q@K^T via WMMA, masks, exp, unnormalized weights store,
//      P@V via WMMA (exp tile re-layout through LDS), rowsum shuffle-reduce,
//      normalize ctx + 16x1024 weight rows
//   3) out projection via WMMA (f16 A from ctx, f16 Wo)
// ---------------------------------------------------------------------------

typedef __attribute__((ext_vector_type(16))) _Float16 v16h;
typedef __attribute__((ext_vector_type(8)))  _Float16 v8h;
typedef __attribute__((ext_vector_type(8)))  float    v8f;

#define EMBED 512
#define NHEAD 8
#define HDIM  64
#define BB    8
#define TT    512
#define SS    1024

static __device__ inline v8f v8f_zero() {
  v8f z;
#pragma unroll
  for (int i = 0; i < 8; ++i) z[i] = 0.0f;
  return z;
}

// D = A*B + C, f16 inputs, f32 accumulate (emits v_wmma_f32_16x16x32_f16)
static __device__ inline v8f wmma32(v16h a, v16h b, v8f c) {
  return __builtin_amdgcn_wmma_f32_16x16x32_f16(
      /*neg_a=*/false, a, /*neg_b=*/false, b,
      /*c_mod=*/(short)0, c, /*reuse_a=*/false, /*reuse_b=*/false);
}

// A fragment (16x32, MxK): halves at row[kOff+0..7] and row[kOff+16..23]
static __device__ inline v16h load_a_f16(const _Float16* row, int kOff) {
  v8h lo = *(const v8h*)(row + kOff);
  v8h hi = *(const v8h*)(row + kOff + 16);
  v16h r;
#pragma unroll
  for (int i = 0; i < 8; ++i) { r[i] = lo[i]; r[i + 8] = hi[i]; }
  return r;
}

static __device__ inline v16h load_a_f32(const float* row, int kOff) {
  v16h r;
#pragma unroll
  for (int i = 0; i < 8; ++i) {
    r[i]     = (_Float16)row[kOff + i];
    r[i + 8] = (_Float16)row[kOff + 16 + i];
  }
  return r;
}

// B fragment (32x16, KxN): halves at row[kOff+0..7] and row[kOff+16..23]
static __device__ inline v16h load_b_f16(const _Float16* row, int kOff) {
  v8h lo = *(const v8h*)(row + kOff);
  v8h hi = *(const v8h*)(row + kOff + 16);
  v16h r;
#pragma unroll
  for (int i = 0; i < 8; ++i) { r[i] = lo[i]; r[i + 8] = hi[i]; }
  return r;
}

// ---------------------------------------------------------------------------
// Elementwise f32 -> f16 convert (for the weight matrices), 8 elems/thread.
// ---------------------------------------------------------------------------
__global__ __launch_bounds__(256) void cvt_f16_kernel(
    const float* __restrict__ src, _Float16* __restrict__ dst, int n) {
  int i = (blockIdx.x * 256 + threadIdx.x) * 8;
  if (i + 8 <= n) {
    float4 a = *(const float4*)(src + i);
    float4 b = *(const float4*)(src + i + 4);
    v8h o;
    o[0] = (_Float16)a.x; o[1] = (_Float16)a.y;
    o[2] = (_Float16)a.z; o[3] = (_Float16)a.w;
    o[4] = (_Float16)b.x; o[5] = (_Float16)b.y;
    o[6] = (_Float16)b.z; o[7] = (_Float16)b.w;
    *(v8h*)(dst + i) = o;
  }
}

// ---------------------------------------------------------------------------
// Projection: Y = X @ W16^T + bias, output f16 in head-split layout.
// MODE 0: Y[((b*H+h)*SEQLEN + t)*64 + d]   (Q, K)
// MODE 1: Y[((b*H+h)*64 + d)*SEQLEN + t]   (V, transposed for P@V B-frags)
// One wave per block computes a 16(M) x 64(N) tile. SEQLEN/MODE are
// compile-time so the epilogue index math folds to shifts/masks.
// ---------------------------------------------------------------------------
template <int SEQLEN, int MODE>
__global__ __launch_bounds__(32) void proj_kernel(
    const float* __restrict__ X, const _Float16* __restrict__ W16,
    const float* __restrict__ bias, _Float16* __restrict__ Y) {
  const int lane = threadIdx.x & 31;
  const int l15 = lane & 15;
  const int hg  = lane >> 4;      // 0 or 1
  const int mBase = blockIdx.x * 16;
  const int nBase = blockIdx.y * 64;

  const float* arow = X + (size_t)(mBase + l15) * EMBED;

  v8f acc[4];
#pragma unroll
  for (int j = 0; j < 4; ++j) acc[j] = v8f_zero();

  for (int k0 = 0; k0 < EMBED; k0 += 32) {
    v16h a = load_a_f32(arow, k0 + (hg << 3));
#pragma unroll
    for (int j = 0; j < 4; ++j) {
      const _Float16* brow = W16 + (size_t)(nBase + j * 16 + l15) * EMBED;
      v16h bf = load_b_f16(brow, k0 + (hg << 3));
      acc[j] = wmma32(a, bf, acc[j]);
    }
  }

#pragma unroll
  for (int j = 0; j < 4; ++j) {
    const int n = nBase + j * 16 + l15;
    const int h = n >> 6, d = n & 63;
    const float bv = bias[n];
#pragma unroll
    for (int v = 0; v < 8; ++v) {
      const int m = mBase + v + (hg << 3);
      const int b = m / SEQLEN, t = m % SEQLEN;   // power-of-two: shift/mask
      const float val = acc[j][v] + bv;
      size_t idx = (MODE == 0)
          ? ((size_t)((b * NHEAD + h) * SEQLEN + t)) * HDIM + d
          : ((size_t)((b * NHEAD + h) * HDIM + d)) * SEQLEN + t;
      Y[idx] = (_Float16)val;
    }
  }
}

// ---------------------------------------------------------------------------
// Attention: one wave per (b, h, 16-row t-tile).
// ---------------------------------------------------------------------------
__global__ __launch_bounds__(32) void attn_kernel(
    const _Float16* __restrict__ Qh, const _Float16* __restrict__ Kh,
    const _Float16* __restrict__ Vt,
    const unsigned char* __restrict__ qpad,
    const unsigned char* __restrict__ kpad,
    const unsigned char* __restrict__ amask,
    float* __restrict__ weights,          // (B,H,T,S) f32
    _Float16* __restrict__ ctx) {         // (B,T,E) f16
  __shared__ _Float16 ldsExp[16 * 32];
  __shared__ float ldsInv[16];

  const int lane = threadIdx.x & 31;
  const int l15 = lane & 15;
  const int hg  = lane >> 4;
  const int tBase = blockIdx.x * 16;
  const int h = blockIdx.y, b = blockIdx.z;
  const size_t bh = (size_t)(b * NHEAD + h);
  const float scale = 0.125f;             // 1/sqrt(64)

  // Q A-fragments for k = 0..31 and 32..63
  const _Float16* qrow = Qh + (bh * TT + tBase + l15) * HDIM;
  const v16h qa0 = load_a_f16(qrow, 0 + (hg << 3));
  const v16h qa1 = load_a_f16(qrow, 32 + (hg << 3));

  v8f cacc[4];
#pragma unroll
  for (int j = 0; j < 4; ++j) cacc[j] = v8f_zero();
  float psum[8];
#pragma unroll
  for (int v = 0; v < 8; ++v) psum[v] = 0.0f;

  for (int sc = 0; sc < SS; sc += 32) {
#pragma unroll
    for (int half = 0; half < 2; ++half) {
      const int s0 = sc + half * 16;
      const _Float16* krow = Kh + (bh * SS + s0 + l15) * HDIM;
      v16h kb0 = load_b_f16(krow, 0 + (hg << 3));
      v16h kb1 = load_b_f16(krow, 32 + (hg << 3));
      v8f sacc = v8f_zero();
      sacc = wmma32(qa0, kb0, sacc);
      sacc = wmma32(qa1, kb1, sacc);

      const int sIdx = s0 + l15;
      const unsigned char kp = kpad[b * SS + sIdx];
#pragma unroll
      for (int v = 0; v < 8; ++v) {
        const int m = v + (hg << 3);
        const int t = tBase + m;
        float sco = sacc[v] * scale;
        if (kp) sco = -1e30f;
        if (!amask[((size_t)(b * TT + t)) * SS + sIdx]) sco = -1e30f;
        const float e = __expf(sco);     // exp(-1e30) flushes to 0
        psum[v] += e;
        weights[(bh * TT + t) * SS + sIdx] = e;        // unnormalized
        ldsExp[m * 32 + half * 16 + l15] = (_Float16)e;
      }
    }
    __syncthreads();
    // exp tile as A-fragment (rows = t, K = 32 s-values)
    v16h ea = load_a_f16(&ldsExp[l15 * 32], (hg << 3));
#pragma unroll
    for (int j = 0; j < 4; ++j) {
      const _Float16* vrow = Vt + (bh * HDIM + j * 16 + l15) * SS + sc;
      v16h vb = load_b_f16(vrow, (hg << 3));
      cacc[j] = wmma32(ea, vb, cacc[j]);
    }
    __syncthreads();
  }

  // Row sums: rows of one tile live entirely in one 16-lane half.
  float invv[8];
#pragma unroll
  for (int v = 0; v < 8; ++v) {
    float s = psum[v];
    s += __shfl_xor(s, 1, 32);
    s += __shfl_xor(s, 2, 32);
    s += __shfl_xor(s, 4, 32);
    s += __shfl_xor(s, 8, 32);
    const int m = v + (hg << 3);
    float inv = (s > 0.0f) ? (1.0f / s) : 0.0f;
    if (qpad[b * TT + tBase + m]) inv = 0.0f;  // zero padded query rows
    invv[v] = inv;
    if (l15 == 0) ldsInv[m] = inv;
  }
  __syncthreads();

  // Normalize context and store as f16 into (B,T,E)
#pragma unroll
  for (int j = 0; j < 4; ++j) {
    const int c = h * HDIM + j * 16 + l15;
#pragma unroll
    for (int v = 0; v < 8; ++v) {
      const int t = tBase + v + (hg << 3);
      ctx[((size_t)(b * TT + t)) * EMBED + c] = (_Float16)(cacc[j][v] * invv[v]);
    }
  }

  // Normalize the 16 x 1024 weight rows (coalesced float4 sweep)
  for (int idx = lane; idx < 16 * (SS / 4); idx += 32) {
    const int m = idx >> 8;                 // / (SS/4)
    const int s4 = idx & ((SS / 4) - 1);
    float4* wp = (float4*)(weights + (bh * TT + tBase + m) * SS) + s4;
    float4 wv = *wp;
    const float scl = ldsInv[m];
    wv.x *= scl; wv.y *= scl; wv.z *= scl; wv.w *= scl;
    *wp = wv;
  }
}

// ---------------------------------------------------------------------------
// Output projection: out = ctx @ Wo^T + bo (f16 A, f16 Wo, f32 out)
// ---------------------------------------------------------------------------
__global__ __launch_bounds__(32) void outproj_kernel(
    const _Float16* __restrict__ ctx, const _Float16* __restrict__ Wo16,
    const float* __restrict__ bo, float* __restrict__ out) {
  const int lane = threadIdx.x & 31;
  const int l15 = lane & 15;
  const int hg  = lane >> 4;
  const int mBase = blockIdx.x * 16;
  const int nBase = blockIdx.y * 64;

  const _Float16* arow = ctx + (size_t)(mBase + l15) * EMBED;

  v8f acc[4];
#pragma unroll
  for (int j = 0; j < 4; ++j) acc[j] = v8f_zero();

  for (int k0 = 0; k0 < EMBED; k0 += 32) {
    v16h a = load_a_f16(arow, k0 + (hg << 3));
#pragma unroll
    for (int j = 0; j < 4; ++j) {
      const _Float16* brow = Wo16 + (size_t)(nBase + j * 16 + l15) * EMBED;
      v16h bf = load_b_f16(brow, k0 + (hg << 3));
      acc[j] = wmma32(a, bf, acc[j]);
    }
  }

#pragma unroll
  for (int j = 0; j < 4; ++j) {
    const int n = nBase + j * 16 + l15;
    const float bv = bo[n];
#pragma unroll
    for (int v = 0; v < 8; ++v) {
      const int m = mBase + v + (hg << 3);
      out[(size_t)m * EMBED + n] = acc[j][v] + bv;
    }
  }
}

// ---------------------------------------------------------------------------
extern "C" void kernel_launch(void* const* d_in, const int* in_sizes, int n_in,
                              void* d_out, int out_size, void* d_ws, size_t ws_size,
                              hipStream_t stream) {
  const float* queries = (const float*)d_in[0];
  const float* keys    = (const float*)d_in[1];
  const float* values  = (const float*)d_in[2];
  const unsigned char* qpad  = (const unsigned char*)d_in[3];
  const unsigned char* kpad  = (const unsigned char*)d_in[4];
  const unsigned char* amask = (const unsigned char*)d_in[5];
  const float* Wq = (const float*)d_in[6];
  const float* bq = (const float*)d_in[7];
  const float* Wk = (const float*)d_in[8];
  const float* bk = (const float*)d_in[9];
  const float* Wv = (const float*)d_in[10];
  const float* bv = (const float*)d_in[11];
  const float* Wo = (const float*)d_in[12];
  const float* bo = (const float*)d_in[13];

  char* ws = (char*)d_ws;
  _Float16* Qh   = (_Float16*)(ws);                          // 4 MB
  _Float16* Kh   = (_Float16*)(ws + ((size_t)4  << 20));     // 8 MB
  _Float16* Vt   = (_Float16*)(ws + ((size_t)12 << 20));     // 8 MB
  _Float16* ctx  = (_Float16*)(ws + ((size_t)20 << 20));     // 4 MB
  _Float16* Wq16 = (_Float16*)(ws + ((size_t)24 << 20));     // 512 KB each
  _Float16* Wk16 = Wq16 + (size_t)EMBED * EMBED;
  _Float16* Wv16 = Wk16 + (size_t)EMBED * EMBED;
  _Float16* Wo16 = Wv16 + (size_t)EMBED * EMBED;

  float* out     = (float*)d_out;                            // (B,T,E)
  float* weights = out + (size_t)BB * TT * EMBED;            // (B,H,T,S)

  // 0) weight conversion: 262144 elems each, 8 per thread
  const int nW = EMBED * EMBED;
  const int cvtBlocks = nW / (256 * 8);
  cvt_f16_kernel<<<cvtBlocks, 256, 0, stream>>>(Wq, Wq16, nW);
  cvt_f16_kernel<<<cvtBlocks, 256, 0, stream>>>(Wk, Wk16, nW);
  cvt_f16_kernel<<<cvtBlocks, 256, 0, stream>>>(Wv, Wv16, nW);
  cvt_f16_kernel<<<cvtBlocks, 256, 0, stream>>>(Wo, Wo16, nW);

  // 1) projections: Q rows = B*T = 4096 ; K,V rows = B*S = 8192
  proj_kernel<TT, 0><<<dim3((BB * TT) / 16, EMBED / 64), 32, 0, stream>>>(
      queries, Wq16, bq, Qh);
  proj_kernel<SS, 0><<<dim3((BB * SS) / 16, EMBED / 64), 32, 0, stream>>>(
      keys, Wk16, bk, Kh);
  proj_kernel<SS, 1><<<dim3((BB * SS) / 16, EMBED / 64), 32, 0, stream>>>(
      values, Wv16, bv, Vt);

  // 2) attention
  attn_kernel<<<dim3(TT / 16, NHEAD, BB), 32, 0, stream>>>(
      Qh, Kh, Vt, qpad, kpad, amask, weights, ctx);

  // 3) output projection
  outproj_kernel<<<dim3((BB * TT) / 16, EMBED / 64), 32, 0, stream>>>(
      ctx, Wo16, bo, out);
}
